// EfficientCrossAttentionHead_489626271899
// MI455X (gfx1250) — matrix operand order repaired and checked by reading, not compiled, not run on hardware
//
#include <hip/hip_runtime.h>
#include <hip/hip_bf16.h>

#define Bn 4
#define Nn 16384
#define Cn 256
#define Hn 64
#define CHUNKS 16

typedef __attribute__((ext_vector_type(16))) __bf16 v16bf;
typedef __attribute__((ext_vector_type(2)))  __bf16 bf16x2;
typedef __attribute__((ext_vector_type(8)))  float  v8f;

union Frag16 { unsigned u[8]; v16bf v; };

__device__ __forceinline__ unsigned short f2bf(float f) {
    __bf16 h = (__bf16)f;                     // native cvt, RNE
    return __builtin_bit_cast(unsigned short, h);
}
__device__ __forceinline__ unsigned packbf2(float lo, float hi) {
    bf16x2 v = { (__bf16)lo, (__bf16)hi };    // native packed cvt
    return __builtin_bit_cast(unsigned, v);
}

// ---------------------------------------------------------------------------
// Kernel 0a: zero the accumulators (ksum[2*B*H], Wctx[2*B*H*H])
// ---------------------------------------------------------------------------
__global__ void zero_acc_kernel(float* __restrict__ ksum, float* __restrict__ Wctx) {
    int i = blockIdx.x * blockDim.x + threadIdx.x;
    if (i < 2 * Bn * Hn) ksum[i] = 0.f;
    if (i < 2 * Bn * Hn * Hn) Wctx[i] = 0.f;
}

// ---------------------------------------------------------------------------
// Kernel 0b: one-shot weight conversion f32 -> bf16.
//   Wbf layout: [6][H][C] in order k0,k1,q0,q1,v0,v1 (row-major, c contig)
// ---------------------------------------------------------------------------
__global__ void wconv_kernel(const float* __restrict__ Wk0, const float* __restrict__ Wk1,
                             const float* __restrict__ Wq0, const float* __restrict__ Wq1,
                             const float* __restrict__ Wv0, const float* __restrict__ Wv1,
                             unsigned short* __restrict__ Wbf)
{
    const int i = blockIdx.x * blockDim.x + threadIdx.x;
    const int SZ = Hn * Cn;
    if (i >= SZ) return;
    Wbf[0 * SZ + i] = f2bf(Wk0[i]);
    Wbf[1 * SZ + i] = f2bf(Wk1[i]);
    Wbf[2 * SZ + i] = f2bf(Wq0[i]);
    Wbf[3 * SZ + i] = f2bf(Wq1[i]);
    Wbf[4 * SZ + i] = f2bf(Wv0[i]);
    Wbf[5 * SZ + i] = f2bf(Wv1[i]);
}

// ---------------------------------------------------------------------------
// Kernel 1: projections (k,q,v) + exp/softmax staging.
//   grid (N/128, B, 2), 256 thr = 8 waves; wave handles one 16-token tile.
//   Outputs: EKt = exp(k)^T  bf16 [2][B][H][N]   (h-major -> ctx A-frags)
//            Vt  = v^T       bf16 [2][B][H][N]   (d-major -> ctx B-frags)
//            Qb  = softmax(q) bf16 [2][B][N][H]  (t-major -> attn A-frags)
//            ksum += column sums of exp(k)        f32 [2][B][H]
// ---------------------------------------------------------------------------
__global__ __launch_bounds__(256)
void proj_softmax_kernel(const float* __restrict__ x0, const float* __restrict__ x1,
                         const unsigned short* __restrict__ Wbf,
                         unsigned short* __restrict__ EKt,
                         unsigned short* __restrict__ Qb,
                         unsigned short* __restrict__ Vt,
                         float* __restrict__ ksum)
{
    const int p    = blockIdx.z;
    const int b    = blockIdx.y;
    const int wave = threadIdx.x >> 5;
    const int lane = threadIdx.x & 31;
    const int half = lane >> 4;
    const int l16  = lane & 15;
    const int row0 = blockIdx.x * 128 + wave * 16;

    const int SZ = Hn * Cn;
    const float* x = p ? x1 : x0;
    const unsigned short* Wk = Wbf + (0 + p) * SZ;
    const unsigned short* Wq = Wbf + (2 + p) * SZ;
    const unsigned short* Wv = Wbf + (4 + p) * SZ;

    const float* xrow = x + ((size_t)b * Nn + row0 + l16) * Cn;

    int kj[8];
    #pragma unroll
    for (int j = 0; j < 8; ++j)
        kj[j] = 2 * j + 16 * half;

    v8f zero = {};
    v8f ck[4], cq[4], cv[4];
    #pragma unroll
    for (int ct = 0; ct < 4; ++ct) { ck[ct] = zero; cq[ct] = zero; cv[ct] = zero; }

    for (int kb = 0; kb < Cn; kb += 32) {
        Frag16 a;
        #pragma unroll
        for (int j = 0; j < 8; ++j)
            a.u[j] = packbf2(xrow[kb + kj[j]], xrow[kb + kj[j] + 1]);

        #pragma unroll
        for (int ct = 0; ct < 4; ++ct) {
            const int n  = ct * 16 + l16;        // weight row == output column
            const int kw = kb + 16 * half;       // even, dword aligned
            const unsigned short* wk = Wk + (size_t)n * Cn + kw;
            const unsigned short* wq = Wq + (size_t)n * Cn + kw;
            const unsigned short* wv = Wv + (size_t)n * Cn + kw;
            Frag16 bk, bq, bv;
            #pragma unroll
            for (int j = 0; j < 8; ++j) {
                bk.u[j] = *(const unsigned*)(wk + 2 * j);
                bq.u[j] = *(const unsigned*)(wq + 2 * j);
                bv.u[j] = *(const unsigned*)(wv + 2 * j);
            }
            ck[ct] = __builtin_amdgcn_wmma_f32_16x16x32_bf16(false, a.v, false, bk.v, (short)0, ck[ct], false, false);
            cq[ct] = __builtin_amdgcn_wmma_f32_16x16x32_bf16(false, a.v, false, bq.v, (short)0, cq[ct], false, false);
            cv[ct] = __builtin_amdgcn_wmma_f32_16x16x32_bf16(false, a.v, false, bv.v, (short)0, cv[ct], false, false);
        }
    }

    const int pb = p * Bn + b;

    // ---- k path: e = exp(logit); store E^T (h-major); accumulate column sums
    #pragma unroll
    for (int ct = 0; ct < 4; ++ct) {
        const int h = ct * 16 + l16;
        unsigned short* ep = EKt + ((size_t)pb * Hn + h) * Nn + row0 + 8 * half;
        float lsum = 0.f;
        #pragma unroll
        for (int r = 0; r < 8; ++r) {
            float e = __expf(ck[ct][r]);
            lsum += e;
            ep[r] = f2bf(e);
        }
        atomicAdd(&ksum[pb * Hn + h], lsum);
    }

    // ---- q path: softmax over the 64 head dims of each token row
    #pragma unroll
    for (int r = 0; r < 8; ++r) {
        float e[4];
        float s = 0.f;
        #pragma unroll
        for (int ct = 0; ct < 4; ++ct) { e[ct] = __expf(cq[ct][r]); s += e[ct]; }
        s += __shfl_xor(s, 1, 32);
        s += __shfl_xor(s, 2, 32);
        s += __shfl_xor(s, 4, 32);
        s += __shfl_xor(s, 8, 32);      // reduction stays within each 16-lane half
        const float inv = 1.f / s;
        const int t = row0 + r + 8 * half;
        unsigned short* qp = Qb + ((size_t)pb * Nn + t) * Hn + l16;
        #pragma unroll
        for (int ct = 0; ct < 4; ++ct)
            qp[ct * 16] = f2bf(e[ct] * inv);
    }

    // ---- v path: store transposed (d-major), contiguous per lane
    #pragma unroll
    for (int ct = 0; ct < 4; ++ct) {
        const int d = ct * 16 + l16;
        unsigned short* vp = Vt + ((size_t)pb * Hn + d) * Nn + row0 + 8 * half;
        #pragma unroll
        for (int r = 0; r < 8; ++r)
            vp[r] = f2bf(cv[ct][r]);
    }
}

// ---------------------------------------------------------------------------
// Kernel 2: context matrices  Wctx[p][b] += E^T(64 x Nchunk) @ V(Nchunk x 64)
//   grid (CHUNKS, B, 2), 128 thr = 4 waves; wave = one 16-row (h) tile.
//   Both operands are token-contiguous per lane -> dword/b128 loads only.
// ---------------------------------------------------------------------------
__global__ __launch_bounds__(128)
void ctx_kernel(const unsigned short* __restrict__ EKt,
                const unsigned short* __restrict__ Vt,
                float* __restrict__ Wctx)
{
    const int p = blockIdx.z, b = blockIdx.y;
    const int wave = threadIdx.x >> 5, lane = threadIdx.x & 31;
    const int half = lane >> 4, l16 = lane & 15;
    const int hrow0 = wave * 16;
    const int t0 = blockIdx.x * (Nn / CHUNKS);
    const int pb = p * Bn + b;

    const unsigned short* E = EKt + (size_t)pb * Hn * Nn + (size_t)(hrow0 + l16) * Nn;
    const unsigned short* V = Vt + (size_t)pb * Hn * Nn;

    int kj[8];
    #pragma unroll
    for (int j = 0; j < 8; ++j)
        kj[j] = 2 * j + 16 * half;

    v8f zero = {};
    v8f c[4];
    #pragma unroll
    for (int ct = 0; ct < 4; ++ct) c[ct] = zero;

    for (int kb = t0; kb < t0 + Nn / CHUNKS; kb += 32) {
        Frag16 a;
        #pragma unroll
        for (int j = 0; j < 8; ++j)
            a.u[j] = *(const unsigned*)(E + kb + kj[j]);   // bf16 pair, dword-aligned

        #pragma unroll
        for (int ct = 0; ct < 4; ++ct) {
            const int d = ct * 16 + l16;
            const unsigned short* vp = V + (size_t)d * Nn + kb + 16 * half;
            Frag16 bb;
            #pragma unroll
            for (int j = 0; j < 8; ++j)
                bb.u[j] = *(const unsigned*)(vp + 2 * j);  // contiguous dwords
            c[ct] = __builtin_amdgcn_wmma_f32_16x16x32_bf16(false, a.v, false, bb.v, (short)0, c[ct], false, false);
        }
    }

    float* W = Wctx + (size_t)pb * Hn * Hn;
    #pragma unroll
    for (int ct = 0; ct < 4; ++ct) {
        #pragma unroll
        for (int r = 0; r < 8; ++r)
            atomicAdd(&W[(size_t)(hrow0 + r + 8 * half) * Hn + ct * 16 + l16], c[ct][r]);
    }
}

// ---------------------------------------------------------------------------
// Kernel 3: normalize context by k column sums, emit bf16 TRANSPOSED:
//   Wnt[pb][d][h] = Wctx[pb][h][d] / ksum[pb][h]   (h-contiguous for attn B)
// ---------------------------------------------------------------------------
__global__ void norm_kernel(const float* __restrict__ Wctx, const float* __restrict__ ksum,
                            unsigned short* __restrict__ Wnt)
{
    int i = blockIdx.x * blockDim.x + threadIdx.x;     // output index
    if (i >= 2 * Bn * Hn * Hn) return;
    int h  = i & (Hn - 1);
    int d  = (i >> 6) & (Hn - 1);
    int pb = i >> 12;
    Wnt[i] = f2bf(Wctx[((size_t)pb * Hn + h) * Hn + d] / ksum[pb * Hn + h]);
}

// ---------------------------------------------------------------------------
// Kernel 4: attn_p = Q_p @ W_{1-p}   (cross pairing), f32 out
//   grid (N/128, B, 2), 256 thr = 8 waves; wave = one 16-token tile.
// ---------------------------------------------------------------------------
__global__ __launch_bounds__(256)
void attn_kernel(const unsigned short* __restrict__ Qb,
                 const unsigned short* __restrict__ Wnt,
                 float* __restrict__ out)
{
    const int p = blockIdx.z, b = blockIdx.y;
    const int wave = threadIdx.x >> 5, lane = threadIdx.x & 31;
    const int half = lane >> 4, l16 = lane & 15;
    const int row0 = blockIdx.x * 128 + wave * 16;
    const int pb  = p * Bn + b;
    const int pbW = (1 - p) * Bn + b;      // cross attention

    const unsigned short* Q = Qb + (size_t)pb * Nn * Hn + (size_t)(row0 + l16) * Hn;
    const unsigned short* W = Wnt + (size_t)pbW * Hn * Hn;   // [d][h], h contig

    int kj[8];
    #pragma unroll
    for (int j = 0; j < 8; ++j)
        kj[j] = 2 * j + 16 * half;

    v8f zero = {};
    v8f c[4];
    #pragma unroll
    for (int ct = 0; ct < 4; ++ct) c[ct] = zero;

    #pragma unroll
    for (int kb = 0; kb < Hn; kb += 32) {
        Frag16 a;
        #pragma unroll
        for (int j = 0; j < 8; ++j)
            a.u[j] = *(const unsigned*)(Q + kb + kj[j]);

        #pragma unroll
        for (int ct = 0; ct < 4; ++ct) {
            const int d = ct * 16 + l16;
            const unsigned short* wp = W + (size_t)d * Hn + kb + 16 * half;
            Frag16 bb;
            #pragma unroll
            for (int j = 0; j < 8; ++j)
                bb.u[j] = *(const unsigned*)(wp + 2 * j);  // contiguous dwords
            c[ct] = __builtin_amdgcn_wmma_f32_16x16x32_bf16(false, a.v, false, bb.v, (short)0, c[ct], false, false);
        }
    }

    float* o = out + (size_t)p * Bn * Nn * Hn + ((size_t)b * Nn + row0) * Hn;
    #pragma unroll
    for (int ct = 0; ct < 4; ++ct) {
        #pragma unroll
        for (int r = 0; r < 8; ++r)
            o[(size_t)(r + 8 * half) * Hn + ct * 16 + l16] = c[ct][r];
    }
}

// ---------------------------------------------------------------------------
extern "C" void kernel_launch(void* const* d_in, const int* in_sizes, int n_in,
                              void* d_out, int out_size, void* d_ws, size_t ws_size,
                              hipStream_t stream)
{
    (void)in_sizes; (void)n_in; (void)out_size; (void)ws_size;

    const float* x0  = (const float*)d_in[0];
    const float* x1  = (const float*)d_in[1];
    const float* Wk0 = (const float*)d_in[2];
    const float* Wk1 = (const float*)d_in[3];
    const float* Wq0 = (const float*)d_in[4];
    const float* Wq1 = (const float*)d_in[5];
    const float* Wv0 = (const float*)d_in[6];
    const float* Wv1 = (const float*)d_in[7];
    float* out = (float*)d_out;

    char* ws = (char*)d_ws;
    const size_t szBF = (size_t)2 * Bn * Nn * Hn * sizeof(unsigned short);  // 16 MB each
    unsigned short* EKt  = (unsigned short*)(ws);
    unsigned short* Qb   = (unsigned short*)(ws + szBF);
    unsigned short* Vt   = (unsigned short*)(ws + 2 * szBF);
    float*          Wctx = (float*)(ws + 3 * szBF);
    unsigned short* Wnt  = (unsigned short*)(ws + 3 * szBF + (size_t)2 * Bn * Hn * Hn * sizeof(float));
    float*          ksum = (float*)(ws + 3 * szBF + (size_t)2 * Bn * Hn * Hn * (sizeof(float) + sizeof(unsigned short)));
    unsigned short* Wbf  = (unsigned short*)(ws + 3 * szBF
                                             + (size_t)2 * Bn * Hn * Hn * (sizeof(float) + sizeof(unsigned short))
                                             + (size_t)2 * Bn * Hn * sizeof(float));

    zero_acc_kernel<<<(2 * Bn * Hn * Hn + 255) / 256, 256, 0, stream>>>(ksum, Wctx);

    wconv_kernel<<<(Hn * Cn + 255) / 256, 256, 0, stream>>>(Wk0, Wk1, Wq0, Wq1, Wv0, Wv1, Wbf);

    proj_softmax_kernel<<<dim3(Nn / 128, Bn, 2), 256, 0, stream>>>(
        x0, x1, Wbf, EKt, Qb, Vt, ksum);

    ctx_kernel<<<dim3(CHUNKS, Bn, 2), 128, 0, stream>>>(EKt, Vt, Wctx);

    norm_kernel<<<(2 * Bn * Hn * Hn + 255) / 256, 256, 0, stream>>>(Wctx, ksum, Wnt);

    attn_kernel<<<dim3(Nn / 128, Bn, 2), 256, 0, stream>>>(Qb, Wnt, out);
}
